// GnnActor_84585085928080
// MI455X (gfx1250) — matrix operand order repaired and checked by reading, not compiled, not run on hardware
//
#include <hip/hip_runtime.h>
#include <hip/hip_bf16.h>
#include <stdint.h>

// ---------------- problem constants (from reference) ----------------
#define BATCH    32768
#define NOBJ     5
#define OBSROW   85          // 10 body + 5*15 obj
#define BM       64          // batch rows per block
#define NTHREADS 128         // 4 waves of 32
// weight fragment buffers in d_ws (dword offsets)
#define W1F_OFF  0           // 9 ktiles * 16 ntiles * 32 lanes * 8 dwords
#define W1F_DW   (9*16*32*8)     // 36864
#define W2F_OFF  (W1F_OFF + W1F_DW)
#define W2F_DW   (8*16*32*8)     // 32768
#define W1RF_OFF (W2F_OFF + W2F_DW)
#define W1RF_DW  (8*16*32*8)
#define TOTAL_FRAG_DW (W1F_DW + W2F_DW + W1RF_DW)  // 102400

typedef __attribute__((ext_vector_type(16))) __bf16 v16bf;
typedef __attribute__((ext_vector_type(8)))  float  v8f;

union Frag { uint4 q[2]; v16bf v; };

__device__ __forceinline__ uint16_t f2bf(float f) {
    uint32_t u = __float_as_uint(f);
    u += 0x7FFFu + ((u >> 16) & 1u);            // round-to-nearest-even
    return (uint16_t)(u >> 16);
}
__device__ __forceinline__ uint32_t pk2bf(float a, float b) {
    return (uint32_t)f2bf(a) | ((uint32_t)f2bf(b) << 16);
}
__device__ __forceinline__ float bf2f(uint16_t h) {
    return __uint_as_float(((uint32_t)h) << 16);
}

// Zero that the compiler cannot see through: redefined by a volatile asm each
// time it executes, so any load address derived from it cannot be hoisted by
// LICM -- but plain pointer arithmetic keeps the global address space intact
// (global_load_b128, not flat_load_b128).
__device__ __forceinline__ int opaque_zero() {
    int x = 0;
    asm volatile("" : "+v"(x));
    return x;
}

// A fragment (16x32 bf16, ISA layout) gathered from row-major bf16 LDS.
// lane m (<16): row m0+m, K = kt*32 + {0..7} in q[0], {16..23} in q[1]
// lane m+16   : row m0+m, K = kt*32 + {8..15} in q[0], {24..31} in q[1]
__device__ __forceinline__ Frag load_afrag(const uint16_t* base, int ldk,
                                           int m0, int kt, int lane) {
    Frag f;
    int m    = m0 + (lane & 15);
    int koff = (lane & 16) ? 8 : 0;
    const uint16_t* p = base + m * ldk + kt * 32 + koff;
    f.q[0] = *(const uint4*)(p);        // 8 bf16 contiguous -> ds_load_b128
    f.q[1] = *(const uint4*)(p + 16);
    return f;
}

// B fragment: pre-swizzled in global ws; per-lane 8 contiguous dwords.
__device__ __forceinline__ Frag load_bfrag(const uint32_t* wf, int kt, int nt, int lane) {
    Frag f;
    const uint4* p = (const uint4*)(wf + ((((kt * 16 + nt) * 32) + lane) << 3));
    f.q[0] = p[0];
    f.q[1] = p[1];
    return f;
}

__device__ __forceinline__ v8f wmma_bf16(Frag a, Frag b, v8f c) {
    return __builtin_amdgcn_wmma_f32_16x16x32_bf16(
        /*neg_a=*/false, a.v, /*neg_b=*/false, b.v,
        /*c_mod=*/(short)0, c, /*reuse_a=*/false, /*reuse_b=*/false);
}

// ---------------- kernel 0: weight pre-swizzle to bf16 B fragments ----------------
// Permuted K order for layer 1: kp<256 -> edge rows (orig 25+kp);
// 256<=kp<281 -> body/obj rows (orig kp-256); kp>=281 -> zero pad.
__device__ __forceinline__ float w1_fetch(const float* __restrict__ w1p, int kp, int n) {
    if (kp < 256) return w1p[(25 + kp) * 256 + n];
    int kl = kp - 256;
    return (kl < 25) ? w1p[kl * 256 + n] : 0.0f;
}

__global__ void prep_weights(const float* __restrict__ w1p,
                             const float* __restrict__ w2p,
                             const float* __restrict__ w1r,
                             uint32_t* __restrict__ ws) {
    int g = blockIdx.x * blockDim.x + threadIdx.x;
    if (g >= TOTAL_FRAG_DW) return;

    const float* W;
    uint32_t* dst;
    int base;
    bool perm = false;
    if (g < W1F_DW)              { W = w1p; dst = ws + W1F_OFF;  base = g;            perm = true; }
    else if (g < W1F_DW + W2F_DW){ W = w2p; dst = ws + W2F_OFF;  base = g - W1F_DW; }
    else                         { W = w1r; dst = ws + W1RF_OFF; base = g - W1F_DW - W2F_DW; }

    int j    = base & 7;
    int lane = (base >> 3) & 31;
    int t    = base >> 8;          // kt*16 + nt
    int nt   = t & 15;
    int kt   = t >> 4;
    int n    = nt * 16 + (lane & 15);
    int k0   = kt * 32 + ((lane & 16) ? 8 : 0) + ((j >= 4) ? 8 : 0) + 2 * j;

    float a, b;
    if (perm) { a = w1_fetch(W, k0, n); b = w1_fetch(W, k0 + 1, n); }
    else      { a = W[k0 * 256 + n];    b = W[(k0 + 1) * 256 + n]; }
    dst[base] = pk2bf(a, b);
}

// ---------------- kernel 1: fully fused GNN actor ----------------
__global__ void __launch_bounds__(NTHREADS, 1)
gnn_actor_fused(const float* __restrict__ obs,
                const float* __restrict__ edge,
                const float* __restrict__ b1p,
                const float* __restrict__ b2p,
                const float* __restrict__ b1r,
                const float* __restrict__ wm, const float* __restrict__ bm,
                const float* __restrict__ wl, const float* __restrict__ bl,
                const uint32_t* __restrict__ ws,
                float* __restrict__ out) {
    __shared__ uint16_t sIn[BM * 256];    // edge_max bf16 / pooled bf16 (A operand)
    __shared__ uint16_t sH[BM * 256];     // h bf16 / x bf16
    __shared__ uint16_t sBO[BM * 32];     // body+obj K-tile (bf16, padded)
    __shared__ float    sPool[BM * 256];  // f32 pooled accumulator

    const uint32_t* w1f  = ws + W1F_OFF;
    const uint32_t* w2f  = ws + W2F_OFF;
    const uint32_t* w1rf = ws + W1RF_OFF;

    const int tid   = threadIdx.x;
    const int lane  = tid & 31;
    const int wave  = tid >> 5;
    const int m0    = wave * 16;                // this wave's 16 rows in tile
    const int b0    = blockIdx.x * BM;          // batch row base
    const int colL  = lane & 15;
    const int rbase = m0 + ((lane >> 4) << 3);  // C-fragment row base (+v = row)

    // ---- zero the pooled accumulator ----
#pragma unroll
    for (int c = 0; c < 32; ++c) {
        int idx = tid + c * NTHREADS;           // 4096 float4 slots
        ((float4*)sPool)[idx] = make_float4(0.f, 0.f, 0.f, 0.f);
    }

    for (int o = 0; o < NOBJ; ++o) {
        __syncthreads();  // protect prior readers of sIn/sH

        // ---- edge max-pool (4 edges) into sIn as bf16, row-major 64x256 ----
        const float* eb = edge + (size_t)(o * 4) * ((size_t)BATCH * 256);
#pragma unroll
        for (int c = 0; c < 32; ++c) {          // 64*64 float4 chunks / 128 thr
            int idx = tid + c * NTHREADS;
            int row = idx >> 6;
            int c4  = idx & 63;
            size_t off = (size_t)(b0 + row) * 256 + (size_t)c4 * 4;
            float4 e0 = *(const float4*)(eb + off);
            float4 e1 = *(const float4*)(eb + (size_t)BATCH * 256 + off);
            float4 e2 = *(const float4*)(eb + (size_t)BATCH * 512 + off);
            float4 e3 = *(const float4*)(eb + (size_t)BATCH * 768 + off);
            float mx = fmaxf(fmaxf(e0.x, e1.x), fmaxf(e2.x, e3.x));
            float my = fmaxf(fmaxf(e0.y, e1.y), fmaxf(e2.y, e3.y));
            float mz = fmaxf(fmaxf(e0.z, e1.z), fmaxf(e2.z, e3.z));
            float mw = fmaxf(fmaxf(e0.w, e1.w), fmaxf(e2.w, e3.w));
            uint32_t* d = (uint32_t*)&sIn[row * 256 + c4 * 4];
            d[0] = pk2bf(mx, my);
            d[1] = pk2bf(mz, mw);
        }
        // ---- body + object K-tile (25 features, padded to 32) ----
#pragma unroll
        for (int c = 0; c < 16; ++c) {          // 64*32 / 128 thr
            int idx = tid + c * NTHREADS;
            int row = idx >> 5;
            int kl  = idx & 31;
            float v = 0.0f;
            if (kl < 10)       v = obs[(size_t)(b0 + row) * OBSROW + kl];
            else if (kl < 25)  v = obs[(size_t)(b0 + row) * OBSROW + 10 + o * 15 + (kl - 10)];
            sBO[row * 32 + kl] = f2bf(v);
        }
        __syncthreads();

        // ---- GEMM1: h = relu(inp @ W1 + b1) -> sH (own rows only) ----
#pragma unroll 2
        for (int nt = 0; nt < 16; ++nt) {
            const uint32_t* bw = w1f + opaque_zero();   // blocks LICM, keeps global AS
            v8f acc = {};
#pragma unroll
            for (int kt = 0; kt < 8; ++kt)
                acc = wmma_bf16(load_afrag(sIn, 256, m0, kt, lane),
                                load_bfrag(bw, kt, nt, lane), acc);
            acc = wmma_bf16(load_afrag(sBO, 32, m0, 0, lane),
                            load_bfrag(bw, 8, nt, lane), acc);
            float bias = b1p[nt * 16 + colL];
            int cc = nt * 16 + colL;
#pragma unroll
            for (int v = 0; v < 8; ++v) {
                float t = acc[v] + bias;
                t = t > 0.0f ? t : 0.0f;
                sH[(rbase + v) * 256 + cc] = f2bf(t);
            }
        }
        __syncthreads();

        // ---- GEMM2: phi = relu(h @ W2 + b2); pooled(LDS) += phi ----
#pragma unroll 2
        for (int nt = 0; nt < 16; ++nt) {
            const uint32_t* bw = w2f + opaque_zero();
            v8f acc = {};
#pragma unroll
            for (int kt = 0; kt < 8; ++kt)
                acc = wmma_bf16(load_afrag(sH, 256, m0, kt, lane),
                                load_bfrag(bw, kt, nt, lane), acc);
            float bias = b2p[nt * 16 + colL];
            int cc = nt * 16 + colL;
#pragma unroll
            for (int v = 0; v < 8; ++v) {
                float t = acc[v] + bias;
                t = t > 0.0f ? t : 0.0f;
                sPool[(rbase + v) * 256 + cc] += t;   // lane-owned slot, race-free
            }
        }
    }

    // ---- rho: x = relu(pooled @ W1r + b1r) ----
    __syncthreads();
#pragma unroll
    for (int c = 0; c < 32; ++c) {              // convert pooled f32 -> bf16 in sIn
        int idx = tid + c * NTHREADS;
        int row = idx >> 6;
        int c4  = idx & 63;
        const float4 p = ((const float4*)sPool)[idx];
        uint32_t* d = (uint32_t*)&sIn[row * 256 + c4 * 4];
        d[0] = pk2bf(p.x, p.y);
        d[1] = pk2bf(p.z, p.w);
    }
    __syncthreads();
#pragma unroll 2
    for (int nt = 0; nt < 16; ++nt) {
        const uint32_t* bw = w1rf + opaque_zero();
        v8f acc = {};
#pragma unroll
        for (int kt = 0; kt < 8; ++kt)
            acc = wmma_bf16(load_afrag(sIn, 256, m0, kt, lane),
                            load_bfrag(bw, kt, nt, lane), acc);
        float bias = b1r[nt * 16 + colL];
        int cc = nt * 16 + colL;
#pragma unroll
        for (int v = 0; v < 8; ++v) {
            float t = acc[v] + bias;
            t = t > 0.0f ? t : 0.0f;
            sH[(rbase + v) * 256 + cc] = f2bf(t);   // x in bf16
        }
    }
    __syncthreads();

    // ---- heads: mean = x@wm+bm ; log_std = clip(x@wl+bl, -20, 2) ----
#pragma unroll
    for (int i = 0; i < 2; ++i) {
        int idx = tid + i * NTHREADS;   // 0..255 -> (row, act)
        int row = idx >> 2;
        int a   = idx & 3;
        float sm = 0.0f, sl = 0.0f;
#pragma unroll 8
        for (int k = 0; k < 256; ++k) {
            float xv = bf2f(sH[row * 256 + k]);
            sm = fmaf(xv, wm[k * 4 + a], sm);
            sl = fmaf(xv, wl[k * 4 + a], sl);
        }
        size_t gb = (size_t)(b0 + row) * 4 + a;
        out[gb] = sm + bm[a];
        float L = sl + bl[a];
        out[(size_t)BATCH * 4 + gb] = fminf(fmaxf(L, -20.0f), 2.0f);
    }
}

// ---------------- host entry ----------------
extern "C" void kernel_launch(void* const* d_in, const int* in_sizes, int n_in,
                              void* d_out, int out_size, void* d_ws, size_t ws_size,
                              hipStream_t stream) {
    (void)in_sizes; (void)n_in; (void)out_size; (void)ws_size;
    const float* obs  = (const float*)d_in[0];
    const float* edge = (const float*)d_in[1];
    const float* w1p  = (const float*)d_in[2];
    const float* b1p  = (const float*)d_in[3];
    const float* w2p  = (const float*)d_in[4];
    const float* b2p  = (const float*)d_in[5];
    const float* w1r  = (const float*)d_in[6];
    const float* b1r  = (const float*)d_in[7];
    const float* wm   = (const float*)d_in[8];
    const float* bm   = (const float*)d_in[9];
    const float* wl   = (const float*)d_in[10];
    const float* bl   = (const float*)d_in[11];
    float* out        = (float*)d_out;
    uint32_t* ws      = (uint32_t*)d_ws;

    prep_weights<<<(TOTAL_FRAG_DW + 255) / 256, 256, 0, stream>>>(w1p, w2p, w1r, ws);
    gnn_actor_fused<<<BATCH / BM, NTHREADS, 0, stream>>>(
        obs, edge, b1p, b2p, b1r, wm, bm, wl, bl, ws, out);
}
